// YOLOXHead_42253888258080
// MI455X (gfx1250) — matrix operand order repaired and checked by reading, not compiled, not run on hardware
//
#include <hip/hip_runtime.h>

// ---------------------------------------------------------------------------
// CDNA5 (gfx1250) YOLOX head via bf16 WMMA implicit GEMM.
// wave32; v_wmma_f32_16x16x32_bf16; f32 accumulate.
// Conv kernels use 32x32 per-wave tiles: 4 WMMAs per 8 b128 loads.
// ---------------------------------------------------------------------------

typedef __bf16 bf16;
typedef bf16  v16bf __attribute__((ext_vector_type(16)));
typedef bf16  v8bf  __attribute__((ext_vector_type(8)));
typedef float v8f   __attribute__((ext_vector_type(8)));

#define BN_EPS 1e-5f

// ---------------------------------------------------------------------------
// Weight packing: fold BN, convert to bf16, swizzle into WMMA B-matrix
// register layout.  B (32x16 K x N, 16-bit):
//   lane = ((K%16)>=8 ? 16:0) + N ;  VGPR v<4 holds K = khalf/2 + 2v (+1), v>=4 holds K = 16 + khalf/2 + 2(v-4) (+1)
// Packed storage: dword index = ((kt*Ntiles + nt)*32 + lane)*8 + v
// K global ordering: k = pos*Cin + cin  (pos = ky*3+kx; cin fastest)
// Optional second tensor w2/b2 is appended as extra couts (for reg+obj fuse).
// ---------------------------------------------------------------------------
__global__ __launch_bounds__(256) void pack_weights(
    const float* __restrict__ w,  const float* __restrict__ g,
    const float* __restrict__ bb, const float* __restrict__ m,
    const float* __restrict__ vv,
    const float* __restrict__ w2, const float* __restrict__ b2,
    int Cin, int npos, int Ntiles, int Nact, int Nact2,
    bf16* __restrict__ packed, float* __restrict__ bias)
{
    int Ktiles = (Cin * npos) / 32;
    int total  = Ktiles * Ntiles * 256;           // dwords
    int t = blockIdx.x * blockDim.x + threadIdx.x;

    // bias (BN-folded) written by the first Ntiles*16 threads
    if (t < Ntiles * 16) {
        int cout = t; float bv = 0.f;
        if (cout < Nact) {
            if (g) {
                float s = g[cout] * rsqrtf(vv[cout] + BN_EPS);
                bv = bb[cout] - m[cout] * s;
            } else bv = bb[cout];
        } else if (cout < Nact + Nact2) {
            bv = b2[cout - Nact];
        }
        bias[cout] = bv;
    }
    if (t >= total) return;

    int v    = t & 7;
    int lane = (t >> 3) & 31;
    int tile = t >> 8;
    int nt   = tile % Ntiles;
    int kt   = tile / Ntiles;
    int nn   = lane & 15;
    int cout = nt * 16 + nn;
    int khalf = (lane >= 16) ? 16 : 0;

    float s = 1.f;
    if (cout < Nact && g) s = g[cout] * rsqrtf(vv[cout] + BN_EPS);

    float wv[2] = {0.f, 0.f};
    #pragma unroll
    for (int j = 0; j < 2; ++j) {
        int k   = kt * 32 + (v < 4 ? (khalf >> 1) + 2 * v : 16 + (khalf >> 1) + 2 * (v - 4)) + j;
        int pos = k / Cin, cin = k % Cin;
        float val = 0.f;
        if (cout < Nact)                 val = w[((size_t)cout * Cin + cin) * npos + pos] * s;
        else if (cout < Nact + Nact2)    val = w2[(size_t)(cout - Nact) * Cin + cin];  // 1x1 only
        wv[j] = val;
    }
    packed[(size_t)t * 2 + 0] = (bf16)wv[0];
    packed[(size_t)t * 2 + 1] = (bf16)wv[1];
}

// ---------------------------------------------------------------------------
// fp32 NCHW -> bf16 pixel-major [b][p][c]  (writes coalesced over c)
// ---------------------------------------------------------------------------
__global__ __launch_bounds__(256) void nchw_to_pixmajor(
    const float* __restrict__ in, bf16* __restrict__ out,
    int C, int HW, int total)
{
    int i = blockIdx.x * blockDim.x + threadIdx.x;
    if (i >= total) return;
    int c = i % C, rem = i / C, p = rem % HW, b = rem / HW;
    out[i] = (bf16)in[((size_t)b * C + c) * HW + p];
}

// ---------------------------------------------------------------------------
// A-matrix fragment load (16x32 bf16, wave32 layout):
//   lanes 0-15 : M=lane,  K 0-7 (v0-3) + K 16-23 (v4-7)
//   lanes 16-31: M=lane-16, K 8-15 + K 24-31
// Per lane: two contiguous 16B loads at base+khalf and base+16+khalf.
// ---------------------------------------------------------------------------
__device__ __forceinline__ v16bf load_a_frag(const bf16* __restrict__ p) {
    union { v16bf v; v8bf h[2]; } u;
    u.h[0] = *(const v8bf*)(p);
    u.h[1] = *(const v8bf*)(p + 16);
    return u.v;
}

__device__ __forceinline__ float silu(float z) {
    return z / (1.f + __expf(-z));
}

#define WMMA_BF16(A, B, C) \
    __builtin_amdgcn_wmma_f32_16x16x32_bf16(false, (A), false, (B), (short)0, (C), false, false)

// ---------------------------------------------------------------------------
// Stem GEMM: 1x1 conv Cin->128 + BN + SiLU. A = bf16 [p][Cin] (unpadded),
// out = padded bf16 [b][(H+2)(W+2)][128]. Block = 128 threads = 4 waves;
// tile 32 pixels x 128 couts; wave = 32x32 (2 A frags x 2 B tiles, 4 accums).
// ---------------------------------------------------------------------------
__global__ __launch_bounds__(128) void gemm_stem(
    const bf16* __restrict__ act, const bf16* __restrict__ wpk,
    const float* __restrict__ bias, bf16* __restrict__ outp,
    int Cin, int H, int W)
{
    const int C = 128;
    int HW   = H * W, Wp = W + 2, Hp = H + 2;
    int lane = threadIdx.x & 31, wid = threadIdx.x >> 5;
    int mrow = lane & 15;
    int khalfA = (lane >= 16) ? 8 : 0;
    int p0 = blockIdx.x * 32 + mrow;
    const bf16* a0base = act + (size_t)p0 * Cin + khalfA;
    const bf16* a1base = act + (size_t)(p0 + 16) * Cin + khalfA;

    v8f acc[2][2] = {};
    int nt0 = wid * 2;
    int Ktiles = Cin / 32;
    for (int kt = 0; kt < Ktiles; ++kt) {
        v16bf A0 = load_a_frag(a0base + kt * 32);
        v16bf A1 = load_a_frag(a1base + kt * 32);
        v16bf B0 = *(const v16bf*)(wpk + (((size_t)kt * 8 + nt0    ) * 32 + lane) * 16);
        v16bf B1 = *(const v16bf*)(wpk + (((size_t)kt * 8 + nt0 + 1) * 32 + lane) * 16);
        acc[0][0] = WMMA_BF16(A0, B0, acc[0][0]);
        acc[0][1] = WMMA_BF16(A0, B1, acc[0][1]);
        acc[1][0] = WMMA_BF16(A1, B0, acc[1][0]);
        acc[1][1] = WMMA_BF16(A1, B1, acc[1][1]);
    }

    int col = lane & 15, rbase = (lane >= 16) ? 8 : 0;
    #pragma unroll
    for (int mi = 0; mi < 2; ++mi) {
        #pragma unroll
        for (int t = 0; t < 2; ++t) {
            int cout = (nt0 + t) * 16 + col;
            float bv = bias[cout];
            #pragma unroll
            for (int r = 0; r < 8; ++r) {
                int p = blockIdx.x * 32 + mi * 16 + rbase + r;
                int b = p / HW, rem = p % HW, y = rem / W, x = rem % W;
                float z = acc[mi][t][r] + bv;
                outp[(((size_t)(b * Hp + y + 1)) * Wp + (x + 1)) * C + cout] = (bf16)silu(z);
            }
        }
    }
}

// ---------------------------------------------------------------------------
// 3x3 conv 128->128 + BN + SiLU, implicit GEMM (K = 9*128 = 1152).
// Input: padded bf16 [b][Hp*Wp][128]. Output padded or unpadded.
// Per wave: 32 pixels x 32 couts; 36 K-steps x 4 WMMA = 144 WMMA.
// Prefetch (global_prefetch_b8) of the next filter position's A rows.
// ---------------------------------------------------------------------------
__global__ __launch_bounds__(128) void gemm_conv3(
    const bf16* __restrict__ actp, const bf16* __restrict__ wpk,
    const float* __restrict__ bias, bf16* __restrict__ out,
    int H, int W, int padOut)
{
    const int C = 128;
    int HW = H * W, Wp = W + 2, Hp = H + 2;
    int lane = threadIdx.x & 31, wid = threadIdx.x >> 5;
    int mrow = lane & 15;
    int khalfA = (lane >= 16) ? 8 : 0;

    int p0 = blockIdx.x * 32 + mrow;
    int b0 = p0 / HW, r0 = p0 % HW, y0 = r0 / W, x0 = r0 % W;
    int p1 = p0 + 16;
    int b1 = p1 / HW, r1 = p1 % HW, y1 = r1 / W, x1 = r1 % W;
    // (ky,kx)=(0,0) corner in padded coords == unpadded (y-1, x-1)
    const bf16* a0base = actp + ((size_t)(b0 * Hp + y0) * Wp + x0) * C + khalfA;
    const bf16* a1base = actp + ((size_t)(b1 * Hp + y1) * Wp + x1) * C + khalfA;

    v8f acc[2][2] = {};
    int nt0 = wid * 2;
    int kt  = 0;
    #pragma unroll
    for (int pos = 0; pos < 9; ++pos) {
        int ky = pos / 3, kx = pos % 3;
        const bf16* ap0 = a0base + ((size_t)ky * Wp + kx) * C;
        const bf16* ap1 = a1base + ((size_t)ky * Wp + kx) * C;
        if (pos < 8) {                       // prefetch next position's A rows
            int kyn = (pos + 1) / 3, kxn = (pos + 1) % 3;
            __builtin_prefetch(a0base + ((size_t)kyn * Wp + kxn) * C, 0, 3);
            __builtin_prefetch(a1base + ((size_t)kyn * Wp + kxn) * C, 0, 3);
        }
        #pragma unroll
        for (int ch = 0; ch < 4; ++ch, ++kt) {
            v16bf A0 = load_a_frag(ap0 + ch * 32);
            v16bf A1 = load_a_frag(ap1 + ch * 32);
            v16bf B0 = *(const v16bf*)(wpk + (((size_t)kt * 8 + nt0    ) * 32 + lane) * 16);
            v16bf B1 = *(const v16bf*)(wpk + (((size_t)kt * 8 + nt0 + 1) * 32 + lane) * 16);
            acc[0][0] = WMMA_BF16(A0, B0, acc[0][0]);
            acc[0][1] = WMMA_BF16(A0, B1, acc[0][1]);
            acc[1][0] = WMMA_BF16(A1, B0, acc[1][0]);
            acc[1][1] = WMMA_BF16(A1, B1, acc[1][1]);
        }
    }

    int col = lane & 15, rbase = (lane >= 16) ? 8 : 0;
    #pragma unroll
    for (int mi = 0; mi < 2; ++mi) {
        #pragma unroll
        for (int t = 0; t < 2; ++t) {
            int cout = (nt0 + t) * 16 + col;
            float bv = bias[cout];
            #pragma unroll
            for (int r = 0; r < 8; ++r) {
                int pp = blockIdx.x * 32 + mi * 16 + rbase + r;
                int bb = pp / HW, rr = pp % HW, yy = rr / W, xx = rr % W;
                float z = acc[mi][t][r] + bv;
                size_t o = padOut ? (((size_t)(bb * Hp + yy + 1)) * Wp + (xx + 1)) * C + cout
                                  : (size_t)pp * C + cout;
                out[o] = (bf16)silu(z);
            }
        }
    }
}

// ---------------------------------------------------------------------------
// cls_pred: 1x1 conv 128->80 + sigmoid, fused write into out[..., 5+c].
// One wave per 16-pixel tile; 5 N-tiles x 4 K-tiles = 20 WMMA.
// ---------------------------------------------------------------------------
__global__ __launch_bounds__(32) void gemm_cls_pred(
    const bf16* __restrict__ act, const bf16* __restrict__ wpk,
    const float* __restrict__ bias, float* __restrict__ out,
    int HW, int aoff)
{
    const int C = 128;
    int lane = threadIdx.x & 31;
    int mrow = lane & 15;
    int p = blockIdx.x * 16 + mrow;
    int khalfA = (lane >= 16) ? 8 : 0;
    const bf16* abase = act + (size_t)p * C + khalfA;

    v8f acc[5] = {};
    for (int kt = 0; kt < 4; ++kt) {
        v16bf a = load_a_frag(abase + kt * 32);
        #pragma unroll
        for (int nt = 0; nt < 5; ++nt) {
            v16bf bm = *(const v16bf*)(wpk + (((size_t)kt * 5 + nt) * 32 + lane) * 16);
            acc[nt] = WMMA_BF16(a, bm, acc[nt]);
        }
    }

    int col = lane & 15, rbase = (lane >= 16) ? 8 : 0;
    #pragma unroll
    for (int nt = 0; nt < 5; ++nt) {
        int cout = nt * 16 + col;
        float bv = bias[cout];
        #pragma unroll
        for (int r = 0; r < 8; ++r) {
            int pp = blockIdx.x * 16 + rbase + r;
            int bb = pp / HW, rr = pp % HW;
            float z = acc[nt][r] + bv;
            out[((size_t)bb * 8400 + aoff + rr) * 85 + 5 + cout] = 1.f / (1.f + __expf(-z));
        }
    }
}

// ---------------------------------------------------------------------------
// reg+obj pred: fused 1x1 conv 128->5 (couts 0-3 reg, 4 obj; padded to 16),
// with box decode: ch0=(v+x)*s, ch1=(v+y)*s, ch2/3=exp(v)*s, ch4=sigmoid(v).
// ---------------------------------------------------------------------------
__global__ __launch_bounds__(32) void gemm_regobj_pred(
    const bf16* __restrict__ act, const bf16* __restrict__ wpk,
    const float* __restrict__ bias, float* __restrict__ out,
    int H, int W, int aoff, float stride_)
{
    const int C = 128;
    int HW = H * W;
    int lane = threadIdx.x & 31;
    int mrow = lane & 15;
    int p = blockIdx.x * 16 + mrow;
    int khalfA = (lane >= 16) ? 8 : 0;
    const bf16* abase = act + (size_t)p * C + khalfA;

    v8f acc = {};
    for (int kt = 0; kt < 4; ++kt) {
        v16bf a  = load_a_frag(abase + kt * 32);
        v16bf bm = *(const v16bf*)(wpk + ((size_t)kt * 32 + lane) * 16);
        acc = WMMA_BF16(a, bm, acc);
    }

    int col = lane & 15, rbase = (lane >= 16) ? 8 : 0;
    if (col < 5) {   // divergence only after all WMMAs
        float bv = bias[col];
        #pragma unroll
        for (int r = 0; r < 8; ++r) {
            int pp = blockIdx.x * 16 + rbase + r;
            int bb = pp / HW, rr = pp % HW, yy = rr / W, xx = rr % W;
            float z = acc[r] + bv;
            float o;
            if      (col == 0) o = (z + (float)xx) * stride_;
            else if (col == 1) o = (z + (float)yy) * stride_;
            else if (col <  4) o = __expf(z) * stride_;
            else               o = 1.f / (1.f + __expf(-z));
            out[((size_t)bb * 8400 + aoff + rr) * 85 + col] = o;
        }
    }
}

// ---------------------------------------------------------------------------
// Host launcher
// ---------------------------------------------------------------------------
extern "C" void kernel_launch(void* const* d_in, const int* in_sizes, int n_in,
                              void* d_out, int out_size, void* d_ws, size_t ws_size,
                              hipStream_t stream)
{
    (void)in_sizes; (void)n_in; (void)out_size;
    struct LvlCfg { int Cin, H, W, aoff; float stride; };
    const LvlCfg Ls[3] = { {128, 80, 80, 0,    8.f},
                           {256, 40, 40, 6400, 16.f},
                           {512, 20, 20, 8000, 32.f} };
    const int B = 8;

    const float* feats[3] = { (const float*)d_in[0], (const float*)d_in[1], (const float*)d_in[2] };
    auto in = [&](int i) { return (const float*)d_in[i]; };

    // --- workspace bump allocator ---
    char* wsb = (char*)d_ws; size_t off = 0;
    auto alloc = [&](size_t bytes) -> void* {
        void* p = wsb + off; off = (off + bytes + 255) & ~(size_t)255; return p;
    };
    bf16* T  = (bf16*)alloc((size_t)B * 6400 * 128 * 2);       // pixel-major input (max level)
    bf16* Ap = (bf16*)alloc((size_t)B * 82 * 82 * 128 * 2);    // stem out (padded)
    bf16* Bp = (bf16*)alloc((size_t)B * 82 * 82 * 128 * 2);    // mid conv out (padded)
    bf16* Cc = (bf16*)alloc((size_t)B * 6400 * 128 * 2);       // branch out (unpadded)

    bf16 *wstem[3], *wc0[3], *wc1[3], *wr0[3], *wr1[3], *wcl[3], *wro[3];
    float *bstem[3], *bc0[3], *bc1[3], *br0[3], *br1[3], *bcl[3], *bro[3];
    for (int l = 0; l < 3; ++l) {
        int kts = Ls[l].Cin / 32;
        wstem[l] = (bf16*)alloc((size_t)kts * 8 * 512 * 2);
        wc0[l]   = (bf16*)alloc((size_t)36 * 8 * 512 * 2);
        wc1[l]   = (bf16*)alloc((size_t)36 * 8 * 512 * 2);
        wr0[l]   = (bf16*)alloc((size_t)36 * 8 * 512 * 2);
        wr1[l]   = (bf16*)alloc((size_t)36 * 8 * 512 * 2);
        wcl[l]   = (bf16*)alloc((size_t)4 * 5 * 512 * 2);
        wro[l]   = (bf16*)alloc((size_t)4 * 1 * 512 * 2);
        bstem[l] = (float*)alloc(128 * 4);
        bc0[l]   = (float*)alloc(128 * 4);
        bc1[l]   = (float*)alloc(128 * 4);
        br0[l]   = (float*)alloc(128 * 4);
        br1[l]   = (float*)alloc(128 * 4);
        bcl[l]   = (float*)alloc(80 * 4);
        bro[l]   = (float*)alloc(16 * 4);
    }
    if (off > ws_size) return;   // workspace too small; bail

    auto pack = [&](const float* w, const float* g, const float* bb, const float* m,
                    const float* vv, const float* w2, const float* b2,
                    int Cin, int npos, int Ntiles, int Nact, int Nact2,
                    bf16* pk, float* bs) {
        int Ktiles = (Cin * npos) / 32;
        int total  = Ktiles * Ntiles * 256;
        pack_weights<<<(total + 255) / 256, 256, 0, stream>>>(
            w, g, bb, m, vv, w2, b2, Cin, npos, Ntiles, Nact, Nact2, pk, bs);
    };

    // --- pack all weights (params flattened: l0..l2, keys sorted per level:
    //     cls0(b,g,m,v,w) cls1 cls_pred(b,w) obj_pred(b,w) reg0 reg1 reg_pred(b,w) stem) ---
    for (int l = 0; l < 3; ++l) {
        int ib = 3 + 31 * l;
        pack(in(ib+30), in(ib+27), in(ib+26), in(ib+28), in(ib+29), nullptr, nullptr,
             Ls[l].Cin, 1, 8, 128, 0, wstem[l], bstem[l]);                        // stem 1x1
        pack(in(ib+4),  in(ib+1),  in(ib+0),  in(ib+2),  in(ib+3),  nullptr, nullptr,
             128, 9, 8, 128, 0, wc0[l], bc0[l]);                                  // cls0 3x3
        pack(in(ib+9),  in(ib+6),  in(ib+5),  in(ib+7),  in(ib+8),  nullptr, nullptr,
             128, 9, 8, 128, 0, wc1[l], bc1[l]);                                  // cls1 3x3
        pack(in(ib+18), in(ib+15), in(ib+14), in(ib+16), in(ib+17), nullptr, nullptr,
             128, 9, 8, 128, 0, wr0[l], br0[l]);                                  // reg0 3x3
        pack(in(ib+23), in(ib+20), in(ib+19), in(ib+21), in(ib+22), nullptr, nullptr,
             128, 9, 8, 128, 0, wr1[l], br1[l]);                                  // reg1 3x3
        pack(in(ib+11), nullptr, in(ib+10), nullptr, nullptr, nullptr, nullptr,
             128, 1, 5, 80, 0, wcl[l], bcl[l]);                                   // cls_pred
        pack(in(ib+25), nullptr, in(ib+24), nullptr, nullptr, in(ib+13), in(ib+12),
             128, 1, 1, 4, 1, wro[l], bro[l]);                                    // reg+obj fused
    }

    // --- per-level pipeline ---
    for (int l = 0; l < 3; ++l) {
        int H = Ls[l].H, W = Ls[l].W, Cin = Ls[l].Cin;
        int HW = H * W;
        size_t padBytes = (size_t)B * (H + 2) * (W + 2) * 128 * 2;
        hipMemsetAsync(Ap, 0, padBytes, stream);   // zero pad rings once per level
        hipMemsetAsync(Bp, 0, padBytes, stream);

        int totalT = B * HW * Cin;
        nchw_to_pixmajor<<<(totalT + 255) / 256, 256, 0, stream>>>(feats[l], T, Cin, HW, totalT);

        int pt32 = (B * HW) / 32;   // 1600 / 400 / 100 — all exact
        int pt16 = (B * HW) / 16;   // 3200 / 800 / 200
        gemm_stem <<<pt32, 128, 0, stream>>>(T,  wstem[l], bstem[l], Ap, Cin, H, W);
        gemm_conv3<<<pt32, 128, 0, stream>>>(Ap, wc0[l],   bc0[l],   Bp, H, W, 1);
        gemm_conv3<<<pt32, 128, 0, stream>>>(Bp, wc1[l],   bc1[l],   Cc, H, W, 0);
        gemm_cls_pred<<<pt16, 32, 0, stream>>>(Cc, wcl[l], bcl[l], (float*)d_out, HW, Ls[l].aoff);
        gemm_conv3<<<pt32, 128, 0, stream>>>(Ap, wr0[l],   br0[l],   Bp, H, W, 1);
        gemm_conv3<<<pt32, 128, 0, stream>>>(Bp, wr1[l],   br1[l],   Cc, H, W, 0);
        gemm_regobj_pred<<<pt16, 32, 0, stream>>>(Cc, wro[l], bro[l], (float*)d_out,
                                                  H, W, Ls[l].aoff, Ls[l].stride);
    }
}
